// RetrievalModel_47656957116745
// MI455X (gfx1250) — compile-verified, matrix operand order checked by reading, not run on hardware
//
#include <hip/hip_runtime.h>
#include <hip/hip_bf16.h>

// ---------------------------------------------------------------------------
// RetrievalModel on MI455X (gfx1250, wave32, WMMA)
//
// Kernels:
//  1) gather:  user_table[user_ids] -> d_out (fp32) and ws f16 matrix [1024x32]
//  2) score:   fused WMMA f16 GEMM (cands x users, K=32) + per-lane reg top-10
//              with group-max early-out so the common path is ~6 VALU per WMMA.
//              grid = (8 user-groups fast, 64 candidate slices slow) so blocks
//              sharing a candidate slice run adjacently -> L2 (192MB) reuse.
//  3) merge:   per-user reduction of 64*2 partial top-10 lists -> 10 indices
// ---------------------------------------------------------------------------

#ifndef __has_builtin
#define __has_builtin(x) 0
#endif

#if defined(__HIP_DEVICE_COMPILE__) && __has_builtin(__builtin_amdgcn_global_load_async_to_lds_b128)
#define HAVE_ASYNC_LDS 1
#else
#define HAVE_ASYNC_LDS 0
#endif

typedef __attribute__((ext_vector_type(16))) _Float16 v16h;
typedef __attribute__((ext_vector_type(8)))  _Float16 v8h;
typedef __attribute__((ext_vector_type(8)))  float    v8f;
typedef __attribute__((ext_vector_type(4)))  float    v4f;
typedef __attribute__((ext_vector_type(4)))  int      v4i;

#define EMB     32
#define BATCH   1024
#define NCAND   500000
#define TOPK    10
#define GROUPS  8          // user groups (blockIdx.x, fast dim for L2 reuse)
#define UPB     128        // users per block = 8 waves * 16
#define SLICES  64         // candidate slices (blockIdx.y)
#define TILE_C  128        // candidates per LDS tile
#define SUBT    (TILE_C / 16)
#define CHUNKS  (TILE_C * EMB / 4 / 256)   // float4 chunks per thread per tile
#define TPS     62         // tiles per slice: ceil(ceil(500000/128)/64)
#define PER_USER (SLICES * 2 * TOPK)       // 1280 partial entries per user

// ---------------------------------------------------------------------------
__global__ __launch_bounds__(256) void gather_kernel(
    const int* __restrict__ ids, const float* __restrict__ table,
    float* __restrict__ out_emb, _Float16* __restrict__ u16)
{
    int e = blockIdx.x * 256 + threadIdx.x;   // 0 .. 32767
    if (e < BATCH * EMB) {
        int b = e >> 5, d = e & 31;
        float v = table[(size_t)ids[b] * EMB + d];
        out_emb[e] = v;            // output #1: user embeddings (fp32)
        u16[e] = (_Float16)v;      // f16 copy for WMMA B operand
    }
}

// ---------------------------------------------------------------------------
__device__ __forceinline__ void wait_async0()
{
#if HAVE_ASYNC_LDS
#if __has_builtin(__builtin_amdgcn_s_wait_asynccnt)
    __builtin_amdgcn_s_wait_asynccnt(0);
#else
    asm volatile("s_wait_asynccnt 0" ::: "memory");
#endif
#endif
}

// Stage one TILE_C x 32 fp32 candidate tile into LDS.
__device__ __forceinline__ void stage_tile(float* dst, const float* src, int rowsValid)
{
    const int tid = threadIdx.x;
    if (rowsValid >= TILE_C) {
#if HAVE_ASYNC_LDS
        #pragma unroll
        for (int j = 0; j < CHUNKS; ++j) {
            int c4 = j * 256 + tid;
            __builtin_amdgcn_global_load_async_to_lds_b128(
                (__attribute__((address_space(1))) v4i*)(src + c4 * 4),
                (__attribute__((address_space(3))) v4i*)(dst + c4 * 4),
                0, 0);
        }
#else
        #pragma unroll
        for (int j = 0; j < CHUNKS; ++j) {
            int c4 = j * 256 + tid;
            *(v4f*)(dst + c4 * 4) = *(const v4f*)(src + c4 * 4);
        }
#endif
    } else {
        // partial last tile: zero-fill out-of-range rows
        #pragma unroll
        for (int j = 0; j < CHUNKS; ++j) {
            int c4  = j * 256 + tid;
            int row = c4 >> 3;                    // 8 float4 per 32-float row
            v4f v = {0.f, 0.f, 0.f, 0.f};
            if (row < rowsValid) v = *(const v4f*)(src + c4 * 4);
            *(v4f*)(dst + c4 * 4) = v;
        }
    }
}

// Insert (s, idx) into an ascending-sorted register top-10 (ts[0] = 10th best).
__device__ __forceinline__ void insert_topk(float (&ts)[TOPK], int (&ti)[TOPK],
                                            float s, int idx)
{
    ts[0] = s; ti[0] = idx;
    #pragma unroll
    for (int j = 0; j < TOPK - 1; ++j) {
        if (ts[j + 1] < ts[j]) {
            float tf = ts[j]; ts[j] = ts[j + 1]; ts[j + 1] = tf;
            int   tn = ti[j]; ti[j] = ti[j + 1]; ti[j + 1] = tn;
        } else break;
    }
}

// ---------------------------------------------------------------------------
__global__ __launch_bounds__(256) void score_topk_kernel(
    const float* __restrict__ cand, const _Float16* __restrict__ u16,
    float* __restrict__ ps, int* __restrict__ pi)
{
    __shared__ float tile[2][TILE_C * EMB];      // 2 x 16KB, double buffered

    const int tid  = threadIdx.x;
    const int wave = tid >> 5;
    const int lane = tid & 31;
    const int col  = lane & 15;                  // user column within wave tile
    const int half = lane >> 4;                  // candidate half (ISA K split)
    const int o    = half * 8;                   // K offset: lanes>=16 hold K 8..15 / 24..31
    const int group = blockIdx.x;                // user group (fast dim)
    const int slice = blockIdx.y;                // candidate slice

    // --- B operand: this wave's 16 users, kept in registers all kernel ---
    // 16-bit B 32x16 layout mirrors A 16x32: lane holds col N=lane%16,
    // halves [0..7]=K o..o+7, [8..15]=K 16+o..23+o.
    const int user = group * UPB + wave * 16 + col;
    const _Float16* up = u16 + (size_t)user * EMB;
    v8h blo = *(const v8h*)(up + o);
    v8h bhi = *(const v8h*)(up + 16 + o);
    v16h bmat;
    #pragma unroll
    for (int i = 0; i < 8; ++i) { bmat[i] = blo[i]; bmat[8 + i] = bhi[i]; }

    // --- per-lane top-10 (sorted ascending, ts[0] = current 10th best) ---
    float ts[TOPK]; int ti[TOPK];
    #pragma unroll
    for (int j = 0; j < TOPK; ++j) { ts[j] = -INFINITY; ti[j] = -1; }

    const int t0 = slice * TPS;
    int ntiles = TPS;
    {
        int rem = NCAND - t0 * TILE_C;
        int avail = (rem > 0) ? (rem + TILE_C - 1) / TILE_C : 0;
        if (avail < ntiles) ntiles = avail;
    }

    if (ntiles > 0)
        stage_tile(tile[0], cand + (size_t)t0 * TILE_C * EMB, NCAND - t0 * TILE_C);

    for (int it = 0; it < ntiles; ++it) {
        const int cur = it & 1;
        wait_async0();            // own async loads for buffer `cur` complete
        __syncthreads();          // all waves' data visible; prev compute done

        if (it + 1 < ntiles) {    // overlap next tile load with compute
            int gt = t0 + it + 1;
            stage_tile(tile[1 - cur], cand + (size_t)gt * TILE_C * EMB,
                       NCAND - gt * TILE_C);
        }
        if (it + 2 < ntiles) {    // warm L2 for the tile after next
            const float* pf = cand + (size_t)(t0 + it + 2) * TILE_C * EMB;
            __builtin_prefetch(pf + tid * 16, 0, 1);
        }

        const int  cbase0    = (t0 + it) * TILE_C;
        const bool tile_full = (cbase0 + TILE_C) <= NCAND;  // uniform branch

        if (tile_full) {
            // ---------------- fast path: no index bound checks ----------------
            #pragma unroll
            for (int t = 0; t < SUBT; ++t) {
                const float* row = &tile[cur][(t * 16 + col) * EMB];
                v4f f0 = *(const v4f*)(row + o);
                v4f f1 = *(const v4f*)(row + o + 4);
                v4f f2 = *(const v4f*)(row + 16 + o);
                v4f f3 = *(const v4f*)(row + 20 + o);
                v16h a;
                #pragma unroll
                for (int i = 0; i < 4; ++i) {
                    a[i]      = (_Float16)f0[i];
                    a[4 + i]  = (_Float16)f1[i];
                    a[8 + i]  = (_Float16)f2[i];
                    a[12 + i] = (_Float16)f3[i];
                }
                v8f c = {};
                c = __builtin_amdgcn_wmma_f32_16x16x32_f16(
                        false, a, false, bmat, (short)0, c, false, false);

                // group-max early out: ~4 v_max3 + 1 cmp in the common case
                float gm = fmaxf(fmaxf(fmaxf(c[0], c[1]), fmaxf(c[2], c[3])),
                                 fmaxf(fmaxf(c[4], c[5]), fmaxf(c[6], c[7])));
                if (gm > ts[0]) {                    // rare: some score enters
                    const int cb = cbase0 + t * 16 + 8 * half;
                    #pragma unroll
                    for (int r = 0; r < 8; ++r)
                        if (c[r] > ts[0]) insert_topk(ts, ti, c[r], cb + r);
                }
            }
        } else {
            // -------------- guarded path: last partial tile only --------------
            #pragma unroll
            for (int t = 0; t < SUBT; ++t) {
                const float* row = &tile[cur][(t * 16 + col) * EMB];
                v4f f0 = *(const v4f*)(row + o);
                v4f f1 = *(const v4f*)(row + o + 4);
                v4f f2 = *(const v4f*)(row + 16 + o);
                v4f f3 = *(const v4f*)(row + 20 + o);
                v16h a;
                #pragma unroll
                for (int i = 0; i < 4; ++i) {
                    a[i]      = (_Float16)f0[i];
                    a[4 + i]  = (_Float16)f1[i];
                    a[8 + i]  = (_Float16)f2[i];
                    a[12 + i] = (_Float16)f3[i];
                }
                v8f c = {};
                c = __builtin_amdgcn_wmma_f32_16x16x32_f16(
                        false, a, false, bmat, (short)0, c, false, false);

                const int cb = cbase0 + t * 16 + 8 * half;
                #pragma unroll
                for (int r = 0; r < 8; ++r) {
                    float s  = c[r];
                    int   ci = cb + r;
                    if (s > ts[0] && ci < NCAND) insert_topk(ts, ti, s, ci);
                }
            }
        }
    }

    // write partial list: [user][slice][half][10]
    size_t obase = (((size_t)user * SLICES + slice) * 2 + half) * TOPK;
    #pragma unroll
    for (int j = 0; j < TOPK; ++j) { ps[obase + j] = ts[j]; pi[obase + j] = ti[j]; }
}

// ---------------------------------------------------------------------------
__global__ __launch_bounds__(256) void topk_merge_kernel(
    const float* __restrict__ ps, const int* __restrict__ pi,
    float* __restrict__ out)
{
    __shared__ float ls[PER_USER];
    __shared__ int   li[PER_USER];
    __shared__ float rs[256];
    __shared__ int   rp[256];

    const int u = blockIdx.x;
    const float* ss = ps + (size_t)u * PER_USER;
    const int*   si = pi + (size_t)u * PER_USER;
    for (int e = threadIdx.x; e < PER_USER; e += 256) { ls[e] = ss[e]; li[e] = si[e]; }
    __syncthreads();

    for (int k = 0; k < TOPK; ++k) {
        float best = -INFINITY; int bpos = -1;
        for (int e = threadIdx.x; e < PER_USER; e += 256) {
            float s = ls[e];
            bool take = (s > best) ||
                        (s == best && bpos >= 0 && li[e] < li[bpos]);
            if (take) { best = s; bpos = e; }
        }
        rs[threadIdx.x] = best; rp[threadIdx.x] = bpos;
        __syncthreads();
        for (int off = 128; off > 0; off >>= 1) {
            if (threadIdx.x < off) {
                float s1 = rs[threadIdx.x],       s2 = rs[threadIdx.x + off];
                int   p1 = rp[threadIdx.x],       p2 = rp[threadIdx.x + off];
                bool take = (p1 < 0 && p2 >= 0) || (s2 > s1) ||
                            (s2 == s1 && p2 >= 0 && p1 >= 0 && li[p2] < li[p1]);
                if (take) { rs[threadIdx.x] = s2; rp[threadIdx.x] = p2; }
            }
            __syncthreads();
        }
        if (threadIdx.x == 0) {
            int pos = rp[0];
            int idx = (pos >= 0) ? li[pos] : -1;
            out[(size_t)BATCH * EMB + (size_t)u * TOPK + k] = (float)idx;
            if (pos >= 0) ls[pos] = -INFINITY;   // remove winner
        }
        __syncthreads();
    }
}

// ---------------------------------------------------------------------------
extern "C" void kernel_launch(void* const* d_in, const int* in_sizes, int n_in,
                              void* d_out, int out_size, void* d_ws, size_t ws_size,
                              hipStream_t stream)
{
    (void)in_sizes; (void)n_in; (void)out_size; (void)ws_size;
    const int*   user_ids   = (const int*)d_in[0];
    const float* user_table = (const float*)d_in[1];
    const float* cand_table = (const float*)d_in[2];
    float* out = (float*)d_out;

    // workspace layout
    _Float16* ws_u16 = (_Float16*)d_ws;                              // 64 KB
    char* p = (char*)d_ws + (size_t)BATCH * EMB * sizeof(_Float16);
    float* ws_ps = (float*)p;                                        // 5.24 MB
    p += (size_t)BATCH * PER_USER * sizeof(float);
    int* ws_pi = (int*)p;                                            // 5.24 MB

    gather_kernel<<<(BATCH * EMB) / 256, 256, 0, stream>>>(
        user_ids, user_table, out, ws_u16);

    score_topk_kernel<<<dim3(GROUPS, SLICES), 256, 0, stream>>>(
        cand_table, ws_u16, ws_ps, ws_pi);

    topk_merge_kernel<<<BATCH, 256, 0, stream>>>(ws_ps, ws_pi, out);
}
